// LinearAttentionBlock_49787260895829
// MI455X (gfx1250) — compile-verified, matrix operand order checked by reading, not run on hardware
//
#include <hip/hip_runtime.h>
#include <hip/hip_bf16.h>
#include <math.h>

// ---------------------------------------------------------------------------
// CDNA5 (gfx1250) bf16 WMMA implementation of a linear-attention block.
// wave32; v_wmma_f32_16x16x32_bf16 for all matrix math; async-to-LDS
// double-buffered GEMM pipeline (ASYNCcnt); 3-pass parallel chunked scan.
// ---------------------------------------------------------------------------

typedef __bf16 bf16_t;
typedef bf16_t v16bf __attribute__((ext_vector_type(16)));
typedef float  v8f   __attribute__((ext_vector_type(8)));

#define WMMA_BF16(a, b, c) \
  __builtin_amdgcn_wmma_f32_16x16x32_bf16(false, (a), false, (b), (short)0, (c), false, false)

// 16-bit A/B fragment loader (16x32 MxK striping, ISA 7.12.2):
// lane = (r16, kh); caller passes  base_ptr + kh*8 ; runs at +0..7 and +16..23.
__device__ __forceinline__ v16bf frag_bf(const bf16_t* p) {
  v16bf r;
#pragma unroll
  for (int j = 0; j < 8; ++j) { r[j] = p[j]; r[j + 8] = p[16 + j]; }
  return r;
}

// Async copy of a 128x64-half tile (row = 64 halfs = 128B) from global
// (row stride ldg halfs) into LDS at byte offset lds_bytes. 256 threads x
// 4 x b128. Tracked by ASYNCcnt.
__device__ __forceinline__ void async_tile_load64(unsigned lds_bytes,
                                                  const bf16_t* g, int ldg,
                                                  int tid) {
  const int s0  = tid * 4;       // 8 segs/row -> 4 consecutive segs, same row
  const int row = s0 >> 3;
  const int seg = s0 & 7;
  const bf16_t* gp = g + (size_t)row * ldg + seg * 8;
  const unsigned lp = lds_bytes + (unsigned)(row * 128 + seg * 16);
#pragma unroll
  for (int j = 0; j < 4; ++j) {
    asm volatile("global_load_async_to_lds_b128 %0, %1, off"
                 :: "v"(lp + 16u * j), "v"(gp + 8 * j)
                 : "memory");
  }
}

// Async copy of a 128x128-half tile (row = 128 halfs = 256B), 256 threads x
// 8 x b128 (each thread one half-row).
__device__ __forceinline__ void async_tile_load128(unsigned lds_bytes,
                                                   const bf16_t* g, int ldg,
                                                   int tid) {
  const int s0  = tid * 8;       // 16 segs/row -> 8 consecutive segs, same row
  const int row = s0 >> 4;
  const int seg = s0 & 15;
  const bf16_t* gp = g + (size_t)row * ldg + seg * 8;
  const unsigned lp = lds_bytes + (unsigned)(row * 256 + seg * 16);
#pragma unroll
  for (int j = 0; j < 8; ++j) {
    asm volatile("global_load_async_to_lds_b128 %0, %1, off"
                 :: "v"(lp + 16u * j), "v"(gp + 8 * j)
                 : "memory");
  }
}

// ---------------------------------------------------------------------------
// Weight convert + transpose: fp32 W[K,N] -> bf16 Wt[N,K]
// ---------------------------------------------------------------------------
__global__ __launch_bounds__(256) void convT_kernel(const float* __restrict__ W,
                                                    bf16_t* __restrict__ Wt,
                                                    int K, int N) {
  size_t idx = (size_t)blockIdx.x * 256 + threadIdx.x;
  int kk = (int)(idx / N);
  int nn = (int)(idx % N);
  Wt[(size_t)nn * K + kk] = (bf16_t)W[idx];
}

// ---------------------------------------------------------------------------
// RMSNorm over C columns, fp32 in -> bf16 out
// ---------------------------------------------------------------------------
__global__ __launch_bounds__(256) void rmsnorm_kernel(const float* __restrict__ x,
                                                      const float* __restrict__ w,
                                                      bf16_t* __restrict__ out, int C) {
  __shared__ float red[8];
  __shared__ float sscale;
  const int row = blockIdx.x;
  const float* px = x + (size_t)row * C;
  float ss = 0.f;
  for (int c = threadIdx.x; c < C; c += 256) { float t = px[c]; ss += t * t; }
  for (int off = 16; off > 0; off >>= 1) ss += __shfl_xor(ss, off, 32);
  if ((threadIdx.x & 31) == 0) red[threadIdx.x >> 5] = ss;
  __syncthreads();
  if (threadIdx.x == 0) {
    float t = 0.f;
    for (int i = 0; i < 8; ++i) t += red[i];
    sscale = rsqrtf(t / (float)C + 1e-5f);
  }
  __syncthreads();
  const float r = sscale;
  bf16_t* po = out + (size_t)row * C;
  for (int c = threadIdx.x; c < C; c += 256) po[c] = (bf16_t)(px[c] * r * w[c]);
}

// per-(token,head) RMSNorm over DV=128 ; o fp32 [BT,1024] -> bf16 [BT,1024]
__global__ __launch_bounds__(256) void headnorm_kernel(const float* __restrict__ o,
                                                       const float* __restrict__ w,
                                                       bf16_t* __restrict__ out) {
  const int row  = blockIdx.x;
  const int h    = threadIdx.x >> 5;
  const int lane = threadIdx.x & 31;
  const float* p = o + (size_t)row * 1024 + h * 128;
  float ss = 0.f;
  for (int c = lane; c < 128; c += 32) { float t = p[c]; ss += t * t; }
  for (int off = 16; off > 0; off >>= 1) ss += __shfl_xor(ss, off, 32);
  const float r = rsqrtf(ss / 128.f + 1e-5f);
  bf16_t* po = out + (size_t)row * 1024 + h * 128;
  for (int c = lane; c < 128; c += 32) po[c] = (bf16_t)(p[c] * r * w[c]);
}

// ---------------------------------------------------------------------------
// Async double-buffered bf16 WMMA GEMM: C[M,N] = A[M,K] * B[K,N].
// Bt (and Bu for GATEUP) are B^T row-major. Block tile 128x128, K-block 64,
// staged via global_load_async_to_lds_b128; 8 waves, each 32x64.
// mode 0: write bf16 ; mode 1: write fp32 + residual ; GATEUP: SiLU(g)*u bf16
// ---------------------------------------------------------------------------
template <bool GATEUP>
__global__ __launch_bounds__(256) void gemm_lds_kernel(
    const bf16_t* __restrict__ A, const bf16_t* __restrict__ B0,
    const bf16_t* __restrict__ B1, int M, int N, int K,
    bf16_t* __restrict__ outBf, float* __restrict__ outF,
    const float* __restrict__ residual, int mode) {
  (void)M;
  extern __shared__ char smem[];
  constexpr int TILE_H = 128 * 64;        // halfs per staged tile (16 KB)
  constexpr int NT     = GATEUP ? 3 : 2;  // tiles per buffer (A,B[,U])
  bf16_t* sbase = (bf16_t*)smem;
  const unsigned lds0 = (unsigned)(uintptr_t)smem;  // flat->LDS: low 32 bits

  const int tid  = threadIdx.x;
  const int lane = tid & 31;
  const int wave = tid >> 5;
  const int kh   = lane >> 4;
  const int r16  = lane & 15;
  const int mw   = wave & 3;   // 4 waves along M (32 rows each)
  const int nw   = wave >> 2;  // 2 waves along N (64 cols each)
  const int m0   = blockIdx.y * 128;
  const int n0   = blockIdx.x * 128;

  const bf16_t* gA = A + (size_t)m0 * K;
  const bf16_t* gB = B0 + (size_t)n0 * K;
  const bf16_t* gU = GATEUP ? (B1 + (size_t)n0 * K) : nullptr;

  auto issue = [&](int buf, int kb) {
    const unsigned lb = lds0 + (unsigned)(buf * NT * TILE_H * 2);
    async_tile_load64(lb, gA + kb * 64, K, tid);
    async_tile_load64(lb + TILE_H * 2, gB + kb * 64, K, tid);
    if (GATEUP) async_tile_load64(lb + 2 * TILE_H * 2, gU + kb * 64, K, tid);
  };

  v8f acc[2][4]  = {};
  v8f accu[2][4] = {};

  const int nb = K / 64;
  issue(0, 0);
  for (int kb = 0; kb < nb; ++kb) {
    const int cur = kb & 1;
    if (kb + 1 < nb) {
      issue(cur ^ 1, kb + 1);
      // our outstanding asyncs == next block's loads -> current block landed
      asm volatile("s_wait_asynccnt %0" :: "i"(GATEUP ? 12 : 8));
    } else {
      asm volatile("s_wait_asynccnt 0");
    }
    __syncthreads();

    bf16_t* cA = sbase + cur * NT * TILE_H;
    bf16_t* cB = cA + TILE_H;
    bf16_t* cU = cB + TILE_H;
#pragma unroll
    for (int k0 = 0; k0 < 64; k0 += 32) {
      v16bf af[2];
#pragma unroll
      for (int tm = 0; tm < 2; ++tm)
        af[tm] = frag_bf(cA + (mw * 32 + tm * 16 + r16) * 64 + k0 + kh * 8);
#pragma unroll
      for (int tn = 0; tn < 4; ++tn) {
        v16bf bf = frag_bf(cB + (nw * 64 + tn * 16 + r16) * 64 + k0 + kh * 8);
#pragma unroll
        for (int tm = 0; tm < 2; ++tm)
          acc[tm][tn] = WMMA_BF16(af[tm], bf, acc[tm][tn]);
        if (GATEUP) {
          v16bf uf = frag_bf(cU + (nw * 64 + tn * 16 + r16) * 64 + k0 + kh * 8);
#pragma unroll
          for (int tm = 0; tm < 2; ++tm)
            accu[tm][tn] = WMMA_BF16(af[tm], uf, accu[tm][tn]);
        }
      }
    }
    __syncthreads();
  }

#pragma unroll
  for (int tm = 0; tm < 2; ++tm)
#pragma unroll
    for (int tn = 0; tn < 4; ++tn)
#pragma unroll
      for (int i = 0; i < 8; ++i) {
        const int row = m0 + mw * 32 + tm * 16 + kh * 8 + i;
        const int col = n0 + nw * 64 + tn * 16 + r16;
        const size_t idx = (size_t)row * N + col;
        if (GATEUP) {
          const float g = acc[tm][tn][i];
          const float u = accu[tm][tn][i];
          outBf[idx] = (bf16_t)((g / (1.f + expf(-g))) * u);
        } else if (mode == 0) {
          outBf[idx] = (bf16_t)acc[tm][tn][i];
        } else {
          outF[idx] = acc[tm][tn][i] + residual[idx];
        }
      }
}

// ---------------------------------------------------------------------------
// Linear attention, 3-pass parallel chunked scan.
// q,k,v: bf16 [B*T, 1024] (head h at cols h*128..h*128+127), CHUNK=128.
// Chunk ids: idx = (b*8 + h)*32 + n  (512 total).
// ---------------------------------------------------------------------------
#define CH 128

// Pass A: KVt[idx][e][d] = sum_i v[i,e] * k[i,d]  (per-chunk outer product)
__global__ __launch_bounds__(256) void kv_outer_kernel(
    const bf16_t* __restrict__ k, const bf16_t* __restrict__ v,
    float* __restrict__ KVt) {
  extern __shared__ char smem[];
  bf16_t* kt = (bf16_t*)smem;   // kt[d*CH + i] = k[i][d]
  bf16_t* vt = kt + CH * CH;    // vt[e*CH + i] = v[i][e]

  const int tid  = threadIdx.x;
  const int lane = tid & 31;
  const int wave = tid >> 5;
  const int kh   = lane >> 4;
  const int r16  = lane & 15;
  const int idx = blockIdx.x;
  const int b = idx >> 8, h = (idx >> 5) & 7, n = idx & 31;
  const size_t base = ((size_t)b * 4096 + (size_t)n * 128) * 1024 + (size_t)h * 128;

  for (int t = tid; t < CH * CH; t += 256) {
    const int i = t >> 7, d = t & 127;
    kt[d * CH + i] = k[base + (size_t)i * 1024 + d];
    vt[d * CH + i] = v[base + (size_t)i * 1024 + d];
  }
  __syncthreads();

  v8f acc[8] = {};
  for (int i0 = 0; i0 < 128; i0 += 32) {
    v16bf a = frag_bf(vt + (wave * 16 + r16) * CH + i0 + kh * 8);  // A = v^T rows e
#pragma unroll
    for (int db = 0; db < 8; ++db) {
      const bf16_t* pKt = kt + (db * 16 + r16) * CH + i0 + kh * 8; // B^T = k^T rows d
      acc[db] = WMMA_BF16(a, frag_bf(pKt), acc[db]);
    }
  }
  float* outp = KVt + (size_t)idx * (CH * CH);
#pragma unroll
  for (int db = 0; db < 8; ++db)
#pragma unroll
    for (int i = 0; i < 8; ++i)
      outp[(wave * 16 + kh * 8 + i) * CH + db * 16 + r16] = acc[db][i];
}

// Pass B: per (b,h), exclusive fp32 prefix over 32 chunk outer products,
// written as bf16 Sx[idx][e][d].
__global__ __launch_bounds__(256) void prefix_kernel(const float* __restrict__ KVt,
                                                     bf16_t* __restrict__ Sx) {
  const int bh = blockIdx.x;  // 16
  const size_t hb = (size_t)bh * 32 * (CH * CH);
  float acc[64];
#pragma unroll
  for (int j = 0; j < 64; ++j) acc[j] = 0.f;
  for (int n = 0; n < 32; ++n) {
    const size_t base = hb + (size_t)n * (CH * CH);
#pragma unroll
    for (int j = 0; j < 64; ++j) {
      const int p = threadIdx.x + 256 * j;
      Sx[base + p] = (bf16_t)acc[j];
      acc[j] += KVt[base + p];
    }
  }
}

// Pass C: o = tril(q k^T) v + q S_n ; one WG per chunk.
// LDS: krm (row-major k, async), vt (transposed v), at (masked attn),
// sx (prefix state, async) = 128 KB.
__global__ __launch_bounds__(256) void attn_out_kernel(
    const bf16_t* __restrict__ q, const bf16_t* __restrict__ k,
    const bf16_t* __restrict__ v, const bf16_t* __restrict__ Sx,
    float* __restrict__ o) {
  extern __shared__ char smem[];
  bf16_t* krm = (bf16_t*)smem;    // k row-major [i][d]
  bf16_t* vt  = krm + CH * CH;    // vt[e][i]
  bf16_t* at  = vt + CH * CH;     // masked attn [i][j]
  bf16_t* sx  = at + CH * CH;     // S^T prefix [e][d]
  const unsigned lds0 = (unsigned)(uintptr_t)smem;

  const int tid  = threadIdx.x;
  const int lane = tid & 31;
  const int wave = tid >> 5;
  const int kh   = lane >> 4;
  const int r16  = lane & 15;
  const int idx = blockIdx.x;
  const int b = idx >> 8, h = (idx >> 5) & 7, n = idx & 31;
  const size_t base = ((size_t)b * 4096 + (size_t)n * 128) * 1024 + (size_t)h * 128;

  // async stage: k chunk (row-major) and prefix state
  async_tile_load128(lds0, k + base, 1024, tid);
  async_tile_load128(lds0 + 3u * CH * CH * 2, Sx + (size_t)idx * (CH * CH), 128, tid);
  // manual transposed stage of v (overlaps with async)
  for (int t = tid; t < CH * CH; t += 256) {
    const int i = t >> 7, e = t & 127;
    vt[e * CH + i] = v[base + (size_t)i * 1024 + e];
  }
  asm volatile("s_wait_asynccnt 0");
  __syncthreads();

  // phase 1: attn = tril(q . k^T) ; wave owns i-block = wave
  {
    v8f acc[8] = {};
    const bf16_t* pQ = q + base + (size_t)(wave * 16 + r16) * 1024 + kh * 8;
    for (int d0 = 0; d0 < 128; d0 += 32) {
      v16bf a = frag_bf(pQ + d0);
#pragma unroll
      for (int tj = 0; tj < 8; ++tj) {
        const bf16_t* pK = krm + (tj * 16 + r16) * CH + d0 + kh * 8;
        acc[tj] = WMMA_BF16(a, frag_bf(pK), acc[tj]);
      }
    }
#pragma unroll
    for (int tj = 0; tj < 8; ++tj)
#pragma unroll
      for (int i = 0; i < 8; ++i) {
        const int gi = wave * 16 + kh * 8 + i;
        const int gj = tj * 16 + r16;
        at[gi * CH + gj] = (bf16_t)((gj <= gi) ? acc[tj][i] : 0.f);
      }
  }
  __syncthreads();

  // phase 2: o = attn . v + q . S ; wave owns i-block, all e-blocks
  {
    v8f acc[8] = {};
    const bf16_t* pAt = at + (wave * 16 + r16) * CH + kh * 8;
    for (int j0 = 0; j0 < 128; j0 += 32) {
      v16bf a = frag_bf(pAt + j0);
#pragma unroll
      for (int eb = 0; eb < 8; ++eb) {
        const bf16_t* pV = vt + (eb * 16 + r16) * CH + j0 + kh * 8;
        acc[eb] = WMMA_BF16(a, frag_bf(pV), acc[eb]);
      }
    }
    const bf16_t* pQ = q + base + (size_t)(wave * 16 + r16) * 1024 + kh * 8;
    for (int d0 = 0; d0 < 128; d0 += 32) {
      v16bf a = frag_bf(pQ + d0);
#pragma unroll
      for (int eb = 0; eb < 8; ++eb) {
        const bf16_t* pS = sx + (eb * 16 + r16) * CH + d0 + kh * 8;
        acc[eb] = WMMA_BF16(a, frag_bf(pS), acc[eb]);
      }
    }
#pragma unroll
    for (int eb = 0; eb < 8; ++eb)
#pragma unroll
      for (int i = 0; i < 8; ++i) {
        const int gi = wave * 16 + kh * 8 + i;
        const int ge = eb * 16 + r16;
        o[base + (size_t)gi * 1024 + ge] = acc[eb][i];
      }
  }
}

// ---------------------------------------------------------------------------
// Host-side orchestration
// ---------------------------------------------------------------------------
extern "C" void kernel_launch(void* const* d_in, const int* in_sizes, int n_in,
                              void* d_out, int out_size, void* d_ws, size_t ws_size,
                              hipStream_t stream) {
  (void)in_sizes; (void)n_in; (void)out_size; (void)ws_size;

  const float* x   = (const float*)d_in[0];
  const float* anw = (const float*)d_in[1];
  const float* Wq  = (const float*)d_in[2];
  const float* Wk  = (const float*)d_in[3];
  const float* Wv  = (const float*)d_in[4];
  const float* Wo  = (const float*)d_in[5];
  const float* onw = (const float*)d_in[6];
  const float* mnw = (const float*)d_in[7];
  const float* Wg  = (const float*)d_in[8];
  const float* Wu  = (const float*)d_in[9];
  const float* Wd  = (const float*)d_in[10];
  float* out = (float*)d_out;

  const int BT = 8192, Dm = 1024, IN = 4096;

  char* ws = (char*)d_ws;
  size_t off = 0;
  auto take = [&](size_t bytes) -> char* {
    char* p = ws + off;
    off += (bytes + 255) & ~(size_t)255;
    return p;
  };
  bf16_t* Wq_t  = (bf16_t*)take((size_t)Dm * Dm * 2);
  bf16_t* Wk_t  = (bf16_t*)take((size_t)Dm * Dm * 2);
  bf16_t* Wv_t  = (bf16_t*)take((size_t)Dm * Dm * 2);
  bf16_t* Wo_t  = (bf16_t*)take((size_t)Dm * Dm * 2);
  bf16_t* Wg_t  = (bf16_t*)take((size_t)Dm * IN * 2);
  bf16_t* Wu_t  = (bf16_t*)take((size_t)Dm * IN * 2);
  bf16_t* Wd_t  = (bf16_t*)take((size_t)IN * Dm * 2);
  bf16_t* h_bf  = (bf16_t*)take((size_t)BT * Dm * 2);   // reused for mlp rmsnorm
  bf16_t* q_bf  = (bf16_t*)take((size_t)BT * Dm * 2);
  bf16_t* k_bf  = (bf16_t*)take((size_t)BT * Dm * 2);
  bf16_t* v_bf  = (bf16_t*)take((size_t)BT * Dm * 2);
  float*  o_f   = (float*) take((size_t)BT * Dm * 4);
  bf16_t* onorm = (bf16_t*)take((size_t)BT * Dm * 2);
  float*  x2    = (float*) take((size_t)BT * Dm * 4);
  char*   actR  = take((size_t)BT * IN * 2);            // 64 MB region
  bf16_t* act   = (bf16_t*)actR;
  // KVt (32 MB) + Sx (16 MB) alias the act region (dead before act is written)
  float*  KVt = (float*)actR;
  bf16_t* Sx  = (bf16_t*)(actR + (size_t)512 * CH * CH * 4);

  const dim3 blk(256);

  // 1) weight convert + transpose (fp32 -> bf16, B^T row-major)
  convT_kernel<<<dim3((Dm * Dm) / 256), blk, 0, stream>>>(Wq, Wq_t, Dm, Dm);
  convT_kernel<<<dim3((Dm * Dm) / 256), blk, 0, stream>>>(Wk, Wk_t, Dm, Dm);
  convT_kernel<<<dim3((Dm * Dm) / 256), blk, 0, stream>>>(Wv, Wv_t, Dm, Dm);
  convT_kernel<<<dim3((Dm * Dm) / 256), blk, 0, stream>>>(Wo, Wo_t, Dm, Dm);
  convT_kernel<<<dim3((Dm * IN) / 256), blk, 0, stream>>>(Wg, Wg_t, Dm, IN);
  convT_kernel<<<dim3((Dm * IN) / 256), blk, 0, stream>>>(Wu, Wu_t, Dm, IN);
  convT_kernel<<<dim3((IN * Dm) / 256), blk, 0, stream>>>(Wd, Wd_t, IN, Dm);

  // 2) attn-input RMSNorm
  rmsnorm_kernel<<<dim3(BT), blk, 0, stream>>>(x, anw, h_bf, Dm);

  // 3) Q/K/V projections (bf16 out); 64 KB LDS double buffer
  const size_t smem2 = 2 * 2 * 128 * 64 * 2;  // 65536
  const size_t smem3 = 2 * 3 * 128 * 64 * 2;  // 98304
  const dim3 gD(Dm / 128, BT / 128);
  gemm_lds_kernel<false><<<gD, blk, smem2, stream>>>(h_bf, Wq_t, nullptr, BT, Dm, Dm, q_bf, nullptr, nullptr, 0);
  gemm_lds_kernel<false><<<gD, blk, smem2, stream>>>(h_bf, Wk_t, nullptr, BT, Dm, Dm, k_bf, nullptr, nullptr, 0);
  gemm_lds_kernel<false><<<gD, blk, smem2, stream>>>(h_bf, Wv_t, nullptr, BT, Dm, Dm, v_bf, nullptr, nullptr, 0);

  // 4) linear attention, 3-pass parallel chunked scan
  kv_outer_kernel<<<dim3(512), blk, 2 * CH * CH * 2, stream>>>(k_bf, v_bf, KVt);
  prefix_kernel<<<dim3(16), blk, 0, stream>>>(KVt, Sx);
  attn_out_kernel<<<dim3(512), blk, 4 * CH * CH * 2, stream>>>(q_bf, k_bf, v_bf, Sx, o_f);

  // 5) per-head output RMSNorm
  headnorm_kernel<<<dim3(BT), blk, 0, stream>>>(o_f, onw, onorm);

  // 6) output projection + residual -> x2 (fp32)
  gemm_lds_kernel<false><<<gD, blk, smem2, stream>>>(onorm, Wo_t, nullptr, BT, Dm, Dm, nullptr, x2, x, 1);

  // 7) mlp RMSNorm (reuse h_bf)
  rmsnorm_kernel<<<dim3(BT), blk, 0, stream>>>(x2, mnw, h_bf, Dm);

  // 8) fused gate/up GEMM + SiLU*up -> bf16 act (96 KB LDS)
  gemm_lds_kernel<true><<<dim3(IN / 128, BT / 128), blk, smem3, stream>>>(
      h_bf, Wg_t, Wu_t, BT, IN, Dm, act, nullptr, nullptr, 2);

  // 9) down projection + residual -> d_out (fp32)
  gemm_lds_kernel<false><<<gD, blk, smem2, stream>>>(act, Wd_t, nullptr, BT, Dm, IN, nullptr, out, x2, 1);
}
